// Cross_CPC_AVT_pad_window_77034533421375
// MI455X (gfx1250) — compile-verified, matrix operand order checked by reading, not run on hardware
//
#include <hip/hip_runtime.h>
#include <hip/hip_bf16.h>
#include <stdint.h>

// Problem constants
#define B_   128
#define T_   256
#define E_   256
#define C_   256
#define NW_  254          // T - NPRED - MIN_START
#define NEGV (-1e9f)

typedef __attribute__((ext_vector_type(16))) __bf16 v16bf;
typedef __attribute__((ext_vector_type(8)))  float  v8f;
typedef __attribute__((ext_vector_type(4)))  unsigned int v4u;
typedef __attribute__((ext_vector_type(8)))  int v8i;
typedef __attribute__((ext_vector_type(4)))  int v4i;

union FragU { unsigned int u[8]; v16bf b; };

__device__ __forceinline__ unsigned short f2bf(float f) {
  unsigned int u = __builtin_bit_cast(unsigned int, f);
  u += 0x7FFFu + ((u >> 16) & 1u);          // round-to-nearest-even
  return (unsigned short)(u >> 16);
}

// Load a 16x32 bf16 operand tile (A-layout per CDNA5 ISA; B mirrored n-major,
// k-contiguous). `base` points at (row0,k0) of a row-major matrix whose rows
// are the M (or N) index and columns are K; dstride = dwords per row.
__device__ __forceinline__ v16bf load_frag(const unsigned int* base, int dstride) {
  const int lane = threadIdx.x & 31;
  const int r    = lane & 15;
  const int hi   = lane >> 4;
  FragU f;
#pragma unroll
  for (int v = 0; v < 8; ++v) {
    const int dk = (v & 3) + hi * 4 + ((v >> 2) << 3);   // k-pair index
    f.u[v] = base[r * dstride + dk];
  }
  return f.b;
}

__device__ __forceinline__ v8f wmma_bf16(v16bf a, v16bf b, v8f c) {
  return __builtin_amdgcn_wmma_f32_16x16x32_bf16(false, a, false, b, (short)0, c,
                                                 false, false);
}

__device__ __forceinline__ float sigm(float x) { return 1.0f / (1.0f + __expf(-x)); }

// Async global->LDS copy (ASYNCcnt path). lds_off is the wave-relative LDS
// byte offset; gptr a per-lane global address. 16 bytes per lane.
__device__ __forceinline__ void async_ld_b128(unsigned lds_off, const void* gptr) {
  asm volatile("global_load_async_to_lds_b128 %0, %1, off"
               :: "v"(lds_off), "v"((unsigned long long)(uintptr_t)gptr)
               : "memory");
}
__device__ __forceinline__ void wait_async() {
  asm volatile("s_wait_asynccnt 0" ::: "memory");
}

// Tensor Data Mover: 2D tile load global -> LDS.
// tile_d0 = contiguous elements per row (dwords, data_size=4B),
// tile_d1 = rows, stride_d0 = row stride in elements. Tensor dims == tile dims
// (exact fit, no OOB), LDS receives a packed tile_d1 x tile_d0 block.
__device__ __forceinline__ void tdm_load_2d(unsigned lds_off, const void* gptr,
                                            unsigned tile_d0, unsigned tile_d1,
                                            unsigned stride_d0) {
  unsigned long long ga = (unsigned long long)(uintptr_t)gptr;
  v4u g0;
  g0[0] = 1u;                                        // count=1, user descriptor
  g0[1] = lds_off;                                   // lds_addr
  g0[2] = (unsigned)(ga & 0xFFFFFFFFu);              // global_addr[31:0]
  g0[3] = (unsigned)((ga >> 32) & 0x01FFFFFFu) | (2u << 30);  // addr[56:32]|type=2
  v8i g1;
  g1[0] = 0x20000;                                   // data_size = 4B
  g1[1] = (int)((tile_d0 & 0xFFFFu) << 16);          // tensor_dim0[15:0] @ bit48
  g1[2] = (int)(((tile_d0 >> 16) & 0xFFFFu) | ((tile_d1 & 0xFFFFu) << 16));
  g1[3] = (int)(((tile_d1 >> 16) & 0xFFFFu) | ((tile_d0 & 0xFFFFu) << 16)); // tile_dim0
  g1[4] = (int)(tile_d1 & 0xFFFFu);                  // tile_dim1 (tile_dim2 = 0)
  g1[5] = (int)stride_d0;                            // tensor_dim0_stride[31:0]
  g1[6] = 0;                                         // stride hi / dim1_stride lo
  g1[7] = 0;
  v4i z4 = {0, 0, 0, 0};
  v8i z8 = {0, 0, 0, 0, 0, 0, 0, 0};
  // 6-arg form (clang-23 / therock-10.0 headers flavor)
  __builtin_amdgcn_tensor_load_to_lds(g0, g1, z4, z4, z8, 0);
}

// ---------------------------------------------------------------- convert x
__global__ void k_cvt_x(const float* a, const float* v, const float* t,
                        unsigned int* xbf) {
  const long long NP = (long long)B_ * T_ * E_ / 2;  // dword pairs / tensor
  long long id = (long long)blockIdx.x * blockDim.x + threadIdx.x;
  if (id >= 3 * NP) return;
  int m = (int)(id / NP);
  long long r = id - (long long)m * NP;
  const float* s = (m == 0) ? a : (m == 1) ? v : t;
  unsigned int d = (unsigned int)f2bf(s[2 * r]) |
                   ((unsigned int)f2bf(s[2 * r + 1]) << 16);
  xbf[(long long)m * NP + r] = d;
}

// -------------------------------------------------- convert/pack weights
// Wcat[m]: rows g in [0,1024) = gates (i,f,g,o), cols k in [0,512) = [Wih|Whh]
__global__ void k_cvt_w(const float* aWih, const float* aWhh,
                        const float* vWih, const float* vWhh,
                        const float* tWih, const float* tWhh,
                        const float* aWp, const float* vWp, const float* tWp,
                        unsigned int* Wcat, unsigned int* Wpb) {
  const int WCP = 1024 * 256;  // dwords per modality Wcat
  const int WPP = 256 * 128;   // dwords per modality Wp
  int id = blockIdx.x * blockDim.x + threadIdx.x;
  if (id < 3 * WCP) {
    int m = id / WCP, r = id - m * WCP;
    int g = r >> 8, dp = r & 255, k = dp * 2;
    const float* ih = (m == 0) ? aWih : (m == 1) ? vWih : tWih;
    const float* hh = (m == 0) ? aWhh : (m == 1) ? vWhh : tWhh;
    float x0, x1;
    if (k < 256) { x0 = ih[g * 256 + k];       x1 = ih[g * 256 + k + 1]; }
    else         { x0 = hh[g * 256 + k - 256]; x1 = hh[g * 256 + k - 255]; }
    Wcat[id] = (unsigned int)f2bf(x0) | ((unsigned int)f2bf(x1) << 16);
  } else {
    int q = id - 3 * WCP;
    if (q >= 3 * WPP) return;
    int m = q / WPP, r = q - m * WPP;
    const float* wp = (m == 0) ? aWp : (m == 1) ? vWp : tWp;
    Wpb[m * WPP + r] = (unsigned int)f2bf(wp[2 * r]) |
                       ((unsigned int)f2bf(wp[2 * r + 1]) << 16);
  }
}

// ------------------------------------------------------------- LSTM scan
// grid = 12 (3 modalities x 4 batch tiles of 32), block = 512 (16 waves)
// Per step: g[32,1024] = [x_t|h][32,512] @ Wcat^T via bf16 WMMA.
// x_t staged via Tensor Data Mover; h kept in a separate LDS block.
// GEMM is k-outer with 8 live accumulators (2 M x 4 N tiles per wave) so the
// A-fragments are loaded once per k-step and nothing is hoisted into spills.
__global__ void k_lstm_scan(const unsigned int* xbf, const unsigned int* Wcat,
                            const float* a_bih, const float* a_bhh,
                            const float* v_bih, const float* v_bhh,
                            const float* t_bih, const float* t_bhh,
                            unsigned int* hs) {
  extern __shared__ char smem_raw[];
  unsigned int* Xr = (unsigned int*)smem_raw;            // 32x128 dw (x_t tile)
  unsigned int* Hr = Xr + 32 * 128;                      // 32x128 dw (h tile)
  float* gbuf = (float*)(smem_raw + 32 * 256 * sizeof(unsigned int)); // 32x1024

  const int m  = blockIdx.x >> 2;
  const int b0 = (blockIdx.x & 3) * 32;
  const int tid = threadIdx.x;

  const unsigned int* xm = xbf + (long long)m * ((long long)B_ * T_ * (E_ / 2));
  const unsigned int* Wm = Wcat + m * (1024 * 256);
  unsigned int* hsm = hs + (long long)m * ((long long)T_ * B_ * (C_ / 2));
  const float* bih = (m == 0) ? a_bih : (m == 1) ? v_bih : t_bih;
  const float* bhh = (m == 0) ? a_bhh : (m == 1) ? v_bhh : t_bhh;

  const int tb = tid >> 4;           // local batch row (0..31)
  const int j0 = (tid & 15) * 16;    // hidden col base for elementwise phase
  float cc[16];
  float bI[16], bF[16], bG[16], bO[16];
#pragma unroll
  for (int ii = 0; ii < 16; ++ii) {
    cc[ii] = 0.f;
    int j = j0 + ii;
    bI[ii] = bih[j]       + bhh[j];
    bF[ii] = bih[256 + j] + bhh[256 + j];
    bG[ii] = bih[512 + j] + bhh[512 + j];
    bO[ii] = bih[768 + j] + bhh[768 + j];
  }
  // zero h block (4096 dwords over 512 threads)
#pragma unroll
  for (int ii = 0; ii < 8; ++ii) Hr[tid * 8 + ii] = 0u;
  const int wv = tid >> 5;
  const int lane = tid & 31;

  for (int t = 0; t < T_; ++t) {
    // TDM: DMA x_t tile (32 rows x 128 dwords, row stride T_*128 dwords)
    if (tid < 32) {
      tdm_load_2d(0u, xm + ((long long)b0 * T_ + t) * 128, 128u, 32u,
                  (unsigned)(T_ * 128));
      __builtin_amdgcn_s_wait_tensorcnt(0);
    }
    __syncthreads();

    // gates GEMM: [32,512] x [512,1024]
    {
      v8f acc[8];                      // [0..3] = M-tile 0, [4..7] = M-tile 1
#pragma unroll
      for (int q = 0; q < 8; ++q) acc[q] = (v8f){0, 0, 0, 0, 0, 0, 0, 0};
#pragma unroll 1
      for (int kt = 0; kt < 16; ++kt) {
        const unsigned int* ab = (kt < 8) ? (Xr + kt * 16)
                                          : (Hr + (kt - 8) * 16);
        v16bf a0 = load_frag(ab, 128);
        v16bf a1 = load_frag(ab + 16 * 128, 128);
#pragma unroll
        for (int nt = 0; nt < 4; ++nt) {
          v16bf bf = load_frag(Wm + ((wv * 4 + nt) * 16) * 256 + kt * 16, 256);
          acc[nt]     = wmma_bf16(a0, bf, acc[nt]);
          acc[4 + nt] = wmma_bf16(a1, bf, acc[4 + nt]);
        }
      }
      const int rb = (lane >> 4) * 8;
#pragma unroll
      for (int nt = 0; nt < 4; ++nt) {
        const int col = (wv * 4 + nt) * 16 + (lane & 15);
#pragma unroll
        for (int r = 0; r < 8; ++r) {
          gbuf[(r + rb) * 1024 + col]      = acc[nt][r];
          gbuf[(16 + r + rb) * 1024 + col] = acc[4 + nt][r];
        }
      }
    }
    __syncthreads();

    // elementwise LSTM cell: thread owns (tb, j0..j0+15), c in registers
    unsigned short hp[16];
#pragma unroll
    for (int ii = 0; ii < 16; ++ii) {
      int j = j0 + ii;
      float gi = gbuf[tb * 1024 + j]       + bI[ii];
      float gf = gbuf[tb * 1024 + 256 + j] + bF[ii];
      float gg = gbuf[tb * 1024 + 512 + j] + bG[ii];
      float go = gbuf[tb * 1024 + 768 + j] + bO[ii];
      float cn = sigm(gf) * cc[ii] + sigm(gi) * tanhf(gg);
      cc[ii] = cn;
      hp[ii] = f2bf(sigm(go) * tanhf(cn));
    }
#pragma unroll
    for (int p = 0; p < 8; ++p) {
      unsigned int d = (unsigned int)hp[2 * p] | ((unsigned int)hp[2 * p + 1] << 16);
      Hr[tb * 128 + (j0 >> 1) + p] = d;                                 // next-step h
      hsm[((long long)t * B_ + (b0 + tb)) * 128 + (j0 >> 1) + p] = d;   // ctx out
    }
    __syncthreads();
  }
}

// ------------------------------------------------------------- predictions
// pred[m][w] = hs[m][w+1] @ Wp^T + bp  -> [128,256] bf16
// grid = 3*NW_, block = 256 (8 waves, one M-tile each); k-outer / n-inner,
// 8 live accumulators per wave (no spills).
__global__ void k_pred(const unsigned int* hs, const unsigned int* Wpb,
                       const float* a_bp, const float* v_bp, const float* t_bp,
                       unsigned short* pred) {
  const int m = blockIdx.x / NW_;
  const int w = blockIdx.x - m * NW_;
  const int tid = threadIdx.x;
  const int wv = tid >> 5, lane = tid & 31;
  const unsigned int* hsm = hs + (long long)m * ((long long)T_ * B_ * 128)
                               + (long long)(w + 1) * B_ * 128;
  const unsigned int* Wpm = Wpb + m * (E_ * 128);
  const float* bp = (m == 0) ? a_bp : (m == 1) ? v_bp : t_bp;
  unsigned short* po = pred + (long long)m * NW_ * B_ * E_ + (long long)w * B_ * E_;
  const int mBase = wv * 16;

#pragma unroll 1
  for (int half = 0; half < 2; ++half) {
    v8f acc[8];
#pragma unroll
    for (int q = 0; q < 8; ++q) acc[q] = (v8f){0, 0, 0, 0, 0, 0, 0, 0};
#pragma unroll 1
    for (int kt = 0; kt < 8; ++kt) {
      v16bf a = load_frag(hsm + mBase * 128 + kt * 16, 128);
#pragma unroll
      for (int q = 0; q < 8; ++q) {
        const int n0 = (half * 8 + q) * 16;
        v16bf bf = load_frag(Wpm + n0 * 128 + kt * 16, 128);  // Wp rows = e = N
        acc[q] = wmma_bf16(a, bf, acc[q]);
      }
    }
    const int rb = (lane >> 4) * 8;
#pragma unroll
    for (int q = 0; q < 8; ++q) {
      const int col = (half * 8 + q) * 16 + (lane & 15);
#pragma unroll
      for (int r = 0; r < 8; ++r) {
        int row = mBase + r + rb;
        po[(long long)row * E_ + col] = f2bf(acc[q][r] + bp[col]);
      }
    }
  }
}

// --------------------------------------------------------- pair kernel
// One workgroup per (pair, window). enc and pred tiles (64KB each) are staged
// into LDS with async global->LDS b128 copies, then total = enc @ pred^T via
// WMMA from LDS, then masked log-softmax diag sum + column argmax accuracy.
__global__ void k_pair(const unsigned int* xbf, const unsigned int* pred,
                       const int* lengths, float* nceT, float* cnt) {
  extern __shared__ char smem_raw[];
  unsigned int* X32 = (unsigned int*)smem_raw;     // 128x128 dw enc tile (64KB)
  unsigned int* P32 = X32 + 128 * 128;             // 128x128 dw pred tile (64KB)
  float* tot = (float*)(smem_raw + 2 * 128 * 128 * sizeof(unsigned int)); // 128x129
  float* lse = tot + 128 * 129;
  float* nacc = lse + 128;
  int*   cacc = (int*)(nacc + 1);

  const int pairIdx = blockIdx.x / NW_;
  const int w = blockIdx.x - pairIdx * NW_;
  const int tid = threadIdx.x;
  const int wv = tid >> 5, lane = tid & 31;
  const int encSel[9]  = {0, 0, 1, 1, 2, 2, 0, 1, 2};   // a,v,t = 0,1,2
  const int predSel[9] = {1, 2, 2, 0, 0, 1, 0, 1, 2};
  const int ia = encSel[pairIdx], ip = predSel[pairIdx];

  const unsigned int* xa = xbf + (long long)ia * ((long long)B_ * T_ * 128);
  const unsigned int* pp = pred + (long long)ip * NW_ * B_ * 128
                                + (long long)w * B_ * 128;
  if (tid == 0) { nacc[0] = 0.f; cacc[0] = 0; }

  // async-stage enc rows (strided) and pred rows (contiguous): 16B per lane
  const unsigned Poff = 128 * 128 * 4;   // LDS byte offset of P32
#pragma unroll 1
  for (int ii = 0; ii < 16; ++ii) {
    int idx = ii * 256 + tid;            // one b128 per idx
    int row = idx >> 5, dq = idx & 31;
    async_ld_b128((unsigned)(idx * 16),
                  xa + ((long long)row * T_ + (w + 2)) * 128 + dq * 4);
    async_ld_b128(Poff + (unsigned)(idx * 16), pp + idx * 4);
  }
  wait_async();
  __syncthreads();

  // GEMM from LDS: wave owns M-tile wv; k-outer, 8 live accumulators
  const int mBase = wv * 16;
  v8f acc[8];
#pragma unroll
  for (int q = 0; q < 8; ++q) acc[q] = (v8f){0, 0, 0, 0, 0, 0, 0, 0};
#pragma unroll 1
  for (int kt = 0; kt < 8; ++kt) {
    v16bf a = load_frag(X32 + mBase * 128 + kt * 16, 128);
#pragma unroll
    for (int q = 0; q < 8; ++q) {
      v16bf bf = load_frag(P32 + (q * 16) * 128 + kt * 16, 128);
      acc[q] = wmma_bf16(a, bf, acc[q]);
    }
  }
  const int rb = (lane >> 4) * 8;
#pragma unroll
  for (int q = 0; q < 8; ++q) {
    const int col = q * 16 + (lane & 15);
#pragma unroll
    for (int r = 0; r < 8; ++r)
      tot[(mBase + r + rb) * 129 + col] = acc[q][r];
  }
  __syncthreads();

  const int thr = w + 2;               // mask: lengths > ts + NPRED
  if (tid < 128) {                     // per-row logsumexp + diag NCE
    const int b = tid;
    float mx = -3.0e38f;
    for (int c = 0; c < 128; ++c) {
      float ml = (lengths[c] > thr) ? tot[b * 129 + c] : NEGV;
      mx = fmaxf(mx, ml);
    }
    float s = 0.f;
    for (int c = 0; c < 128; ++c) {
      float ml = (lengths[c] > thr) ? tot[b * 129 + c] : NEGV;
      s += __expf(ml - mx);
    }
    float l = mx + __logf(s);
    lse[b] = l;
    if (lengths[b] > thr) atomicAdd(&nacc[0], tot[b * 129 + b] - l);
  }
  __syncthreads();
  if (tid < 128) {                     // per-column argmax over softmax rows
    const int c = tid;
    if (lengths[c] > thr) {
      float best = -2.0f; int arg = -1;
      for (int b = 0; b < 128; ++b) {
        float smv = (lengths[b] > thr) ? __expf(tot[b * 129 + c] - lse[b]) : -1.0f;
        if (smv > best) { best = smv; arg = b; }
      }
      if (arg == c) atomicAdd(&cacc[0], 1);
    }
  }
  __syncthreads();
  if (tid == 0) {
    nceT[pairIdx * NW_ + w] = nacc[0];
    cnt[pairIdx * NW_ + w]  = (float)cacc[0];
  }
}

// --------------------------------------------------------- final reduce
__global__ void k_final(const float* nceT, const float* cnt,
                        const int* lengths, float* out) {
  __shared__ float s_nce[256];
  __shared__ float s_acc[9][256];
  __shared__ int   s_val[256];
  const int tid = threadIdx.x;
  const float wgt[9] = {1.f, 1.f, 1.f, 1.f, 1.f, 1.f, 0.1f, 0.1f, 0.1f};
  float nv = 0.f; int vf = 0; float av[9];
  for (int j = 0; j < 9; ++j) av[j] = 0.f;
  if (tid < NW_) {
    int vc = 0;
    for (int b = 0; b < B_; ++b) vc += (lengths[b] > tid + 2) ? 1 : 0;
    float safe = (float)((vc > 1) ? vc : 1);
    vf = (vc > 0) ? 1 : 0;
    float nw = 0.f;
    for (int j = 0; j < 9; ++j) {
      nw += wgt[j] * nceT[j * NW_ + tid];
      av[j] = vf ? (cnt[j * NW_ + tid] / safe) : 0.f;
    }
    nv = vf ? (nw / (-safe)) : 0.f;     // NPRED == 1
  }
  s_nce[tid] = nv; s_val[tid] = vf;
  for (int j = 0; j < 9; ++j) s_acc[j][tid] = av[j];
  __syncthreads();
  if (tid == 0) {
    int nvld = 0;
    for (int w2 = 0; w2 < NW_; ++w2) nvld += s_val[w2];
    float nwin = (float)((nvld > 1) ? nvld : 1);
    for (int j = 0; j < 9; ++j) {
      float s = 0.f;
      for (int w2 = 0; w2 < NW_; ++w2) s += s_acc[j][w2];
      out[j] = s / nwin;
    }
    float s = 0.f;
    for (int w2 = 0; w2 < NW_; ++w2) s += s_nce[w2];
    out[9] = s / nwin;
  }
}

// ------------------------------------------------------------ launcher
extern "C" void kernel_launch(void* const* d_in, const int* in_sizes, int n_in,
                              void* d_out, int out_size, void* d_ws, size_t ws_size,
                              hipStream_t stream) {
  (void)in_sizes; (void)n_in; (void)out_size; (void)ws_size;
  const float* audio = (const float*)d_in[0];
  const float* video = (const float*)d_in[1];
  const float* text  = (const float*)d_in[2];
  const float* v_Wih = (const float*)d_in[3];
  const float* v_Whh = (const float*)d_in[4];
  const float* v_bih = (const float*)d_in[5];
  const float* v_bhh = (const float*)d_in[6];
  const float* a_Wih = (const float*)d_in[7];
  const float* a_Whh = (const float*)d_in[8];
  const float* a_bih = (const float*)d_in[9];
  const float* a_bhh = (const float*)d_in[10];
  const float* t_Wih = (const float*)d_in[11];
  const float* t_Whh = (const float*)d_in[12];
  const float* t_bih = (const float*)d_in[13];
  const float* t_bhh = (const float*)d_in[14];
  const float* v_Wp  = (const float*)d_in[15];
  const float* v_bp  = (const float*)d_in[16];
  const float* a_Wp  = (const float*)d_in[17];
  const float* a_bp  = (const float*)d_in[18];
  const float* t_Wp  = (const float*)d_in[19];
  const float* t_bp  = (const float*)d_in[20];
  const int*   lengths = (const int*)d_in[21];

  char* ws = (char*)d_ws;
  size_t off = 0;
  auto alloc = [&](size_t bytes) -> void* {
    void* p = ws + off;
    off += (bytes + 255) & ~(size_t)255;
    return p;
  };
  unsigned int*  xbf  = (unsigned int*)alloc(3ull * B_ * T_ * E_ * 2);     // bf16 x
  unsigned int*  hs   = (unsigned int*)alloc(3ull * T_ * B_ * C_ * 2);     // bf16 ctx [T,B,C]
  unsigned short* pred = (unsigned short*)alloc(3ull * NW_ * B_ * E_ * 2); // bf16
  unsigned int*  Wcat = (unsigned int*)alloc(3ull * 1024 * 512 * 2);       // bf16 [Wih|Whh]
  unsigned int*  Wpb  = (unsigned int*)alloc(3ull * E_ * C_ * 2);          // bf16 Wp
  float* nceT = (float*)alloc(9ull * NW_ * 4);
  float* cnts = (float*)alloc(9ull * NW_ * 4);

  {
    long long np = 3ll * B_ * T_ * E_ / 2;
    int blocks = (int)((np + 255) / 256);
    k_cvt_x<<<blocks, 256, 0, stream>>>(audio, video, text, xbf);
  }
  {
    int total = 3 * 1024 * 256 + 3 * 256 * 128;
    int blocks = (total + 255) / 256;
    k_cvt_w<<<blocks, 256, 0, stream>>>(a_Wih, a_Whh, v_Wih, v_Whh, t_Wih, t_Whh,
                                        a_Wp, v_Wp, t_Wp, Wcat, Wpb);
  }
  {
    size_t shmem = 32 * 256 * sizeof(unsigned int) + 32 * 1024 * sizeof(float); // 160KB
    k_lstm_scan<<<12, 512, shmem, stream>>>(xbf, Wcat, a_bih, a_bhh, v_bih, v_bhh,
                                            t_bih, t_bhh, hs);
  }
  k_pred<<<3 * NW_, 256, 0, stream>>>(hs, Wpb, a_bp, v_bp, t_bp, pred);
  {
    size_t shmem = 2 * 128 * 128 * sizeof(unsigned int)
                 + (128 * 129 + 128 + 2) * sizeof(float) + 16;  // ~195KB
    k_pair<<<9 * NW_, 256, shmem, stream>>>(xbf, (const unsigned int*)pred,
                                            lengths, nceT, cnts);
  }
  k_final<<<1, 256, 0, stream>>>(nceT, cnts, lengths, (float*)d_out);
}
